// CompositionalMlp_19808389169164
// MI455X (gfx1250) — matrix operand order repaired and checked by reading, not compiled, not run on hardware
//
#include <hip/hip_runtime.h>

// MI455X / gfx1250 compositional MoE-MLP.
// Expert-routed GEMM chain using V_WMMA_F32_16X16X32_F16 (fp16 in, fp32 acc).
// 64-row (4 M-tile) blocks; each B fragment feeds 4 WMMAs. Chunked (KTU
// K-tiles) double-buffered LDS staging -> 16 WMMAs per barrier interval.
// sched_group_barrier pipelines DS fragment loads one group ahead of WMMAs.

typedef __attribute__((ext_vector_type(16))) _Float16 v16h;
typedef __attribute__((ext_vector_type(8)))  float    v8f;

#define B_N      8192
#define M_EXP    4
#define IN_COLS  144
#define PAD_ROWS (B_N + M_EXP * 64)      // 8448: expert segments padded to x64
#define NBLK_M   (PAD_ROWS / 64)         // 132 row blocks of 64

#if defined(__has_builtin) && __has_builtin(__builtin_amdgcn_sched_group_barrier)
#define SGB(mask, size) __builtin_amdgcn_sched_group_barrier((mask), (size), 0)
#else
#define SGB(mask, size)
#endif
// masks: 0x008 = MFMA/WMMA, 0x020 = VMEM read, 0x100 = DS read

// ---------------------------------------------------------------------------
// Routing: init scratch, argmax of one-hots, counting sort padded to 64.
// ---------------------------------------------------------------------------
__global__ void k_init(int* __restrict__ perm, int* __restrict__ meta) {
  int tid = blockIdx.x * blockDim.x + threadIdx.x;
  if (tid < 4 * PAD_ROWS) perm[tid] = -1;
  if (tid < 52) meta[tid] = 0;   // counts[16] cursors[16] offs[20]
}

__global__ void k_route(const float* __restrict__ inp, int* __restrict__ idx,
                        int* __restrict__ counts) {
  int b = blockIdx.x * blockDim.x + threadIdx.x;
  if (b >= B_N) return;
  const float* oh = inp + (size_t)b * IN_COLS + 128;
  #pragma unroll
  for (int j = 0; j < 4; ++j) {
    float best = oh[j * 4]; int e = 0;
    #pragma unroll
    for (int q = 1; q < 4; ++q) {
      float v = oh[j * 4 + q];
      if (v > best) { best = v; e = q; }
    }
    idx[j * B_N + b] = e;
    atomicAdd(&counts[j * 4 + e], 1);
  }
}

__global__ void k_offsets(const int* __restrict__ counts, int* __restrict__ offs) {
  if (threadIdx.x | blockIdx.x) return;
  for (int j = 0; j < 4; ++j) {
    int run = 0;
    for (int e = 0; e < 4; ++e) {
      offs[j * 5 + e] = run;
      run += (counts[j * 4 + e] + 63) & ~63;   // pad each expert segment to x64
    }
    offs[j * 5 + 4] = run;
  }
}

__global__ void k_scatter(const int* __restrict__ idx, const int* __restrict__ offs,
                          int* __restrict__ cursors, int* __restrict__ perm) {
  int tid = blockIdx.x * blockDim.x + threadIdx.x;
  if (tid >= 4 * B_N) return;
  int j = tid >> 13, b = tid & (B_N - 1);
  int e = idx[j * B_N + b];
  int pos = offs[j * 5 + e] + atomicAdd(&cursors[j * 4 + e], 1);
  perm[j * PAD_ROWS + pos] = b;
}

// fp32 features -> fp16 (row-major B x 128)
__global__ void k_feats(const float* __restrict__ inp, _Float16* __restrict__ fh) {
  int tid = blockIdx.x * blockDim.x + threadIdx.x;
  if (tid >= B_N * 128) return;
  int b = tid >> 7, c = tid & 127;
  fh[tid] = (_Float16)inp[(size_t)b * IN_COLS + c];
}

// ---------------------------------------------------------------------------
// Weight conversion: fp32 [M][K][Nreal] -> fp16 pre-swizzled WMMA B-fragments
// layout [e][kt][nt][lane(32)][16 halves]; B k order used here (contiguous k = 16h+s; A matches):
//   lane<16: col=lane,    K = 2d, 2d+1        (d = dword 0..7)
//   lane>=16: col=lane-16, K = 16+2d, 17+2d
// ---------------------------------------------------------------------------
__global__ void k_convw(const float* __restrict__ src, _Float16* __restrict__ dst,
                        int K, int Nreal, int Npad) {
  int tid = blockIdx.x * blockDim.x + threadIdx.x;
  int nNT = Npad >> 4, nKT = K >> 5;
  int d    = tid & 7;
  int lane = (tid >> 3) & 31;
  int rest = tid >> 8;
  int nt = rest % nNT; rest /= nNT;
  int kt = rest % nKT;
  int e  = rest / nKT;
  if (e >= M_EXP) return;
  int nloc  = lane & 15;
  int kbase = (lane < 16) ? 0 : 16;
  int n = nt * 16 + nloc;
  size_t fb = (((size_t)(e * nKT + kt)) * nNT + nt) * 512 + lane * 16 + d * 2;
  #pragma unroll
  for (int h = 0; h < 2; ++h) {
    int k = kt * 32 + kbase + 2 * d + h;
    float v = (n < Nreal) ? src[((size_t)e * K + k) * Nreal + n] : 0.f;
    dst[fb + h] = (_Float16)v;
  }
}

// ---------------------------------------------------------------------------
// Expert GEMM: one block = 64 expert-uniform rows (4 M-tiles) x 8 N-tiles
// (one per wave; waves duplicate N-tiles when nNT < 8 -> no divergent guard).
// Per chunk: gather-stage 64 x KTU*32 halves (double buffered), then KTU*4
// WMMAs. Pad rows stage row 0's data; their outputs are masked on store.
// ---------------------------------------------------------------------------
template <int KTU>
__global__ __launch_bounds__(256) void k_gemm(
    const _Float16* __restrict__ A, int lda,          // halves
    const _Float16* __restrict__ W,                   // swizzled fragments
    int K, int nNT,
    const float* __restrict__ bias, int Nreal,
    const int* __restrict__ perm, const int* __restrict__ offs,  // [5]
    _Float16* __restrict__ Oh, int ldo,
    float* __restrict__ Of, int ldof,
    int doRelu) {
  constexpr int RS = KTU * 16 + 4;                    // LDS row stride (dwords)
  __shared__ int sA[2][64 * RS];
  const int row0 = blockIdx.x * 64;
  if (row0 >= offs[4]) return;                        // fully-padded tail block
  int e = 0;
  while (e < 3 && row0 >= offs[e + 1]) ++e;           // expert of this block
  const int nKT = K >> 5;
  const int nCH = nKT / KTU;

  // Staging role: thread -> (row, 8-half column chunk).
  const int sr  = threadIdx.x >> 2;                   // 0..63
  const int scc = threadIdx.x & 3;                    // 0..3
  int sgrow = perm[row0 + sr];
  if (sgrow < 0) sgrow = 0;                           // pad row: any valid data
  const _Float16* aRow = A + (size_t)sgrow * lda + scc * 8;

  #pragma unroll
  for (int u = 0; u < KTU; ++u)
    *(int4*)(&sA[0][sr * RS + u * 16 + scc * 4]) =
        *(const int4*)(aRow + u * 32);
  __syncthreads();

  const int wave = threadIdx.x >> 5;
  const int lane = threadIdx.x & 31;
  const int nt   = (blockIdx.y * 8 + wave) % nNT;     // wave-uniform, no guard

  const int m0    = lane & 15;
  const int abase = (lane < 16) ? 0 : 8;              // dword base in K-tile slice
  const _Float16* wp = W + ((size_t)e * nKT * nNT + nt) * 512 + lane * 16;
  const size_t wstride = (size_t)nNT * 512;

  v8f acc[4] = {};
  int cur = 0;
  for (int kc = 0; kc < nCH; ++kc) {
    const bool more = (kc + 1 < nCH);                 // scalar-uniform
    int4 nx[KTU];
    if (more) {
      #pragma unroll
      for (int u = 0; u < KTU; ++u)
        nx[u] = *(const int4*)(aRow + (size_t)((kc + 1) * KTU + u) * 32);
    }
    #pragma unroll
    for (int u = 0; u < KTU; ++u) {
      const int kt = kc * KTU + u;
      v16h bf = *(const v16h*)(wp + (size_t)kt * wstride);
      union AF { int4 q[2]; v16h h; } af[4];
      #pragma unroll
      for (int mt = 0; mt < 4; ++mt) {
        const int* sb = &sA[cur][(mt * 16 + m0) * RS + u * 16 + abase];
        af[mt].q[0] = *(const int4*)(sb);             // ds_load_b128
        af[mt].q[1] = *(const int4*)(sb + 4);
      }
      #pragma unroll
      for (int mt = 0; mt < 4; ++mt)
        acc[mt] = __builtin_amdgcn_wmma_f32_16x16x32_f16(
            false, af[mt].h, false, bf, (short)0, acc[mt], false, false);
    }
    // Pipelined issue order for this scheduling region: all VMEM reads up
    // front, then DS pairs one group ahead of the WMMA that consumes them.
    SGB(0x020, 3 * KTU);                              // B frags + A prefetch
    SGB(0x100, 4);                                    // 2 DS pairs ahead
    #pragma unroll
    for (int i = 0; i < 4 * KTU - 2; ++i) {
      SGB(0x008, 1);                                  // 1 WMMA
      SGB(0x100, 2);                                  // next DS pair
    }
    SGB(0x008, 2);                                    // trailing WMMAs
    if (more) {
      #pragma unroll
      for (int u = 0; u < KTU; ++u)
        *(int4*)(&sA[cur ^ 1][sr * RS + u * 16 + scc * 4]) = nx[u];
      __syncthreads();
      cur ^= 1;
    }
  }

  // Epilogue: C/D layout -> lane holds col = nt*16+(lane&15), rows rbase+0..7.
  const int col = nt * 16 + m0;
  float bv = (col < Nreal) ? bias[(size_t)e * Nreal + col] : 0.f;
  const int rbase = (lane < 16) ? 0 : 8;
  #pragma unroll
  for (int mt = 0; mt < 4; ++mt) {
    #pragma unroll
    for (int r = 0; r < 8; ++r) {
      int grow = perm[row0 + mt * 16 + rbase + r];
      if (grow < 0 || col >= Nreal) continue;
      float v = acc[mt][r] + bv;
      if (doRelu && v < 0.f) v = 0.f;
      if (Oh) Oh[(size_t)grow * ldo + col] = (_Float16)v;
      else    Of[(size_t)grow * ldof + col] = v;
    }
  }
}

// ---------------------------------------------------------------------------
extern "C" void kernel_launch(void* const* d_in, const int* in_sizes, int n_in,
                              void* d_out, int out_size, void* d_ws, size_t ws_size,
                              hipStream_t stream) {
  (void)in_sizes; (void)n_in; (void)out_size; (void)ws_size;
  const float* inp = (const float*)d_in[0];
  const float* Wsrc[9] = {(const float*)d_in[1],  (const float*)d_in[3],
                          (const float*)d_in[5],  (const float*)d_in[7],
                          (const float*)d_in[9],  (const float*)d_in[11],
                          (const float*)d_in[13], (const float*)d_in[15],
                          (const float*)d_in[17]};
  const float* Bias[9] = {(const float*)d_in[2],  (const float*)d_in[4],
                          (const float*)d_in[6],  (const float*)d_in[8],
                          (const float*)d_in[10], (const float*)d_in[12],
                          (const float*)d_in[14], (const float*)d_in[16],
                          (const float*)d_in[18]};
  // layers:       W0_0 W0_1 W1_0 W1_1 W2_0 W2_1 W3_0 W3_1 W3_2
  const int Ks[9]    = {32, 512, 32, 1024, 32, 1024, 32, 1024, 512};
  const int Nreal[9] = {512, 512, 512, 512, 512, 512, 512, 512, 8};
  const int Npad[9]  = {512, 512, 512, 512, 512, 512, 512, 512, 16};

  // Workspace carve-out (~51 MB total).
  char* p = (char*)d_ws;
  auto alloc = [&](size_t bytes) {
    void* r = (void*)p; p += (bytes + 255) & ~(size_t)255; return r;
  };
  _Float16* Wh[9];
  for (int i = 0; i < 9; ++i)
    Wh[i] = (_Float16*)alloc((size_t)M_EXP * Ks[i] * Npad[i] * 2);
  _Float16* featsH = (_Float16*)alloc((size_t)B_N * 128 * 2);
  _Float16* bufA   = (_Float16*)alloc((size_t)B_N * 1024 * 2);
  _Float16* bufB   = (_Float16*)alloc((size_t)B_N * 1024 * 2);
  int* idx  = (int*)alloc((size_t)4 * B_N * 4);
  int* perm = (int*)alloc((size_t)4 * PAD_ROWS * 4);
  int* meta = (int*)alloc(256);
  int* counts = meta, *cursors = meta + 16, *offs = meta + 32;

  // Phase 0: routing + precision conversion (every launch; no cached state).
  k_init<<<(4 * PAD_ROWS + 255) / 256, 256, 0, stream>>>(perm, meta);
  k_route<<<B_N / 256, 256, 0, stream>>>(inp, idx, counts);
  k_offsets<<<1, 32, 0, stream>>>(counts, offs);
  k_scatter<<<4 * B_N / 256, 256, 0, stream>>>(idx, offs, cursors, perm);
  k_feats<<<B_N * 128 / 256, 256, 0, stream>>>(inp, featsH);
  for (int i = 0; i < 9; ++i) {
    int blocks = M_EXP * (Ks[i] / 32) * (Npad[i] / 16);
    k_convw<<<blocks, 256, 0, stream>>>(Wsrc[i], Wh[i], Ks[i], Nreal[i], Npad[i]);
  }

  // Phase 1: WMMA GEMM chain. bufX cols [0,512)=x, [512,1024)=h (concat order).
  auto gemm = [&](const _Float16* A, int lda, int li, int j,
                  _Float16* Oh, int ldo, float* Of, int ldof, int relu) {
    int nNT = Npad[li] / 16;
    dim3 grid(NBLK_M, (nNT + 7) / 8);
    if (Ks[li] == 32)
      k_gemm<1><<<grid, 256, 0, stream>>>(A, lda, Wh[li], Ks[li], nNT, Bias[li],
                                          Nreal[li], perm + j * PAD_ROWS,
                                          offs + j * 5, Oh, ldo, Of, ldof, relu);
    else
      k_gemm<4><<<grid, 256, 0, stream>>>(A, lda, Wh[li], Ks[li], nNT, Bias[li],
                                          Nreal[li], perm + j * PAD_ROWS,
                                          offs + j * 5, Oh, ldo, Of, ldof, relu);
  };
  gemm(featsH + 0,   128,  0, 0, bufA + 0,   1024, nullptr, 0, 1); // 32  -> x
  gemm(bufA + 0,     1024, 1, 0, bufB + 0,   1024, nullptr, 0, 1); // 512 -> x
  gemm(featsH + 32,  128,  2, 1, bufB + 512, 1024, nullptr, 0, 1); // 32  -> h
  gemm(bufB + 0,     1024, 3, 1, bufA + 0,   1024, nullptr, 0, 1); // 1024-> x
  gemm(featsH + 64,  128,  4, 2, bufA + 512, 1024, nullptr, 0, 1); // 32  -> h
  gemm(bufA + 0,     1024, 5, 2, bufB + 0,   1024, nullptr, 0, 1); // 1024-> x
  gemm(featsH + 96,  128,  6, 3, bufB + 512, 1024, nullptr, 0, 1); // 32  -> h
  gemm(bufB + 0,     1024, 7, 3, bufA + 0,   1024, nullptr, 0, 1); // 1024-> x
  gemm(bufA + 0,     1024, 8, 3, nullptr, 0, (float*)d_out, 8, 0); // head
}
